// MSCAM_11759620457202
// MI455X (gfx1250) — matrix operand order repaired and checked by reading, not compiled, not run on hardware
//
#include <hip/hip_runtime.h>
#include <hip/hip_bf16.h>

typedef __attribute__((ext_vector_type(16))) _Float16 v16h;
typedef __attribute__((ext_vector_type(8)))  _Float16 v8h_t;
typedef __attribute__((ext_vector_type(8)))  float    v8f;
typedef __attribute__((ext_vector_type(4)))  int      i32x4;

#define N_    64
#define CIN   64
#define T_    256
#define V_    25
#define COUT  64
#define NSUB  3
#define MTOT  192      // NSUB*COUT
#define COLS  6400     // T_*V_   (x0 layout)
#define VP    32       // padded V
#define COLSP 8192     // T_*VP   (padded intermediate layout)
#define KNN   5
#define EPS_  1e-5f

#if defined(__has_builtin)
#if __has_builtin(__builtin_amdgcn_global_load_async_to_lds_b128) && \
    __has_builtin(__builtin_amdgcn_s_wait_asynccnt)
#define ASYNC_LDS 1
#endif
#endif

// Build a 16-bit A-matrix fragment (16x32, MxK) per CDNA5 ISA 7.12.2:
// lane l: M = l&15; elems 0..7 -> K = (l>>4)*8 + e ; elems 8..15 -> K = 16 + (l>>4)*8 + e
__device__ __forceinline__ v16h wmma_pack_a(const _Float16* row, int k0, int half) {
  v8h_t lo = *(const v8h_t*)(row + k0 + half * 8);
  v8h_t hi = *(const v8h_t*)(row + k0 + 16 + half * 8);
  v16h a;
#pragma unroll
  for (int e = 0; e < 8; ++e) { a[e] = lo[e]; a[e + 8] = hi[e]; }
  return a;
}

// ---------------- K1: xm[n,c,v] = mean_t x0[n,c,t,v] ----------------
__global__ void k1_mean_t(const float* __restrict__ x0, float* __restrict__ xm) {
  __shared__ float sm[V_];
  int tid = threadIdx.x;
  int nc = blockIdx.x;
  if (tid < V_) sm[tid] = 0.f;
  __syncthreads();
  int lane = tid & 31, wave = tid >> 5;
  if (lane < V_) {
    const float* p = x0 + (size_t)nc * T_ * V_;
    float s = 0.f;
    for (int t = wave; t < T_; t += 8) s += p[t * V_ + lane];
    atomicAdd(&sm[lane], s);
  }
  __syncthreads();
  if (tid < V_) xm[(size_t)nc * V_ + tid] = sm[tid] * (1.0f / T_);
}

// ---------------- K2a: h[n,o,v] = cs_w @ xm + cs_b ----------------
__global__ void k2a_h(const float* __restrict__ xm, const float* __restrict__ cs_w,
                      const float* __restrict__ cs_b, float* __restrict__ h) {
  __shared__ float xl[CIN][V_];
  int tid = threadIdx.x, n = blockIdx.x;
  for (int flat = tid; flat < CIN * V_; flat += 256)
    xl[flat / V_][flat % V_] = xm[(size_t)n * CIN * V_ + flat];
  __syncthreads();
  for (int flat = tid; flat < COUT * V_; flat += 256) {
    int o = flat / V_, v = flat % V_;
    float s = cs_b[o];
    for (int c = 0; c < CIN; ++c) s += xl[c][v] * cs_w[o * CIN + c];
    h[((size_t)n * COUT + o) * V_ + v] = s;
  }
}

// ---------------- K2b: A_corr[v,w] = mean_{n,c} cs_alpha[c/16]*sigmoid(h[n,c,v]-h[n,c,w]) --
__global__ void k2b_acorr(const float* __restrict__ h, const float* __restrict__ cs_alpha,
                          float* __restrict__ acorr) {
  __shared__ float red[256];
  int tid = threadIdx.x;
  int v = blockIdx.x / V_, w = blockIdx.x % V_;
  float s = 0.f;
  for (int flat = tid; flat < N_ * COUT; flat += 256) {
    int c = flat & 63;
    float d = h[(size_t)flat * V_ + v] - h[(size_t)flat * V_ + w];
    s += cs_alpha[c >> 4] / (1.f + __expf(-d));
  }
  red[tid] = s;
  __syncthreads();
  for (int off = 128; off; off >>= 1) {
    if (tid < off) red[tid] += red[tid + off];
    __syncthreads();
  }
  if (tid == 0) acorr[blockIdx.x] = red[0] / (float)(N_ * COUT);
}

// ---------------- K3: x[n,m,t,vp] = conv_w[m,:] . x0[n,:,t,v] + conv_b  (WMMA) ------------
// Output stored V-padded to 32 so downstream tiles are contiguous 128B-aligned blocks.
__global__ void k3_conv_wmma(const float* __restrict__ x0,
                             const float* __restrict__ conv_w,
                             const float* __restrict__ conv_b,
                             _Float16* __restrict__ xws) {
  __shared__ _Float16 Wl[MTOT][64];   // row-major M x K
  __shared__ _Float16 Xt[64][64];     // transposed: [padded col][k]
  const int tid = threadIdx.x;
  const int n = blockIdx.x >> 7;              // 128 col-tiles of 64 padded cols
  const int colBase = (blockIdx.x & 127) * 64;

  for (int flat = tid; flat < MTOT * 64; flat += 256)
    Wl[flat >> 6][flat & 63] = (_Float16)conv_w[flat];

  const float* xbase = x0 + (size_t)n * CIN * COLS;
  for (int flat = tid; flat < 64 * 64; flat += 256) {
    int c = flat >> 6, jj = flat & 63;
    int jp = colBase + jj;
    int t = jp >> 5, v = jp & 31;
    Xt[jj][c] = (v < V_) ? (_Float16)xbase[(size_t)c * COLS + t * V_ + v] : (_Float16)0.f;
  }
  __syncthreads();

  const int wave = tid >> 5, lane = tid & 31;
  const int half = lane >> 4, l16 = lane & 15;
  for (int tile = wave; tile < 48; tile += 8) {   // 12 M-tiles x 4 col-tiles
    int m0 = (tile >> 2) * 16;
    int c0 = (tile & 3) * 16;
    v8f acc;
#pragma unroll
    for (int r = 0; r < 8; ++r) acc[r] = 0.f;
#pragma unroll
    for (int kk = 0; kk < 2; ++kk) {
      int k0 = kk * 32;
      v16h a = wmma_pack_a(&Wl[m0 + l16][0], k0, half);
      v16h b = wmma_pack_a(&Xt[c0 + l16][0], k0, half);
      acc = __builtin_amdgcn_wmma_f32_16x16x32_f16(false, a, false, b, (short)0, acc,
                                                   false, false);
    }
    int col = colBase + c0 + l16;
#pragma unroll
    for (int r = 0; r < 8; ++r) {
      int m = m0 + r + 8 * half;
      xws[((size_t)n * MTOT + m) * COLSP + col] = (_Float16)(acc[r] + conv_b[m]);
    }
  }
}

// ---------------- K4: Q/K pooling over 8 channels x 256 frames ----------------
__global__ void k4_qk(const _Float16* __restrict__ xws, float* __restrict__ Q,
                      float* __restrict__ Kv) {
  __shared__ float sq[V_], sk[V_];
  int tid = threadIdx.x;
  int n = blockIdx.x / 12;
  int rem = blockIdx.x % 12;
  int i = rem / 4, j = rem % 4;
  if (tid < V_) { sq[tid] = 0.f; sk[tid] = 0.f; }
  __syncthreads();
  int v = tid % V_, s8 = tid / V_;
  if (tid < 200) {
    float aq = 0.f, ak = 0.f;
    for (int ch = 0; ch < 16; ++ch) {
      const _Float16* p = xws + ((size_t)n * MTOT + i * 64 + j * 16 + ch) * COLSP + v;
      if (ch < 15)
        __builtin_prefetch(p + COLSP, 0, 1);   // global_prefetch_b8 next channel plane
      float acc = 0.f;
      for (int t = s8; t < T_; t += 8) acc += (float)p[t * VP];
      if (ch < 8) aq += acc; else ak += acc;
    }
    atomicAdd(&sq[v], aq);
    atomicAdd(&sk[v], ak);
  }
  __syncthreads();
  if (tid < V_) {
    size_t base = (((size_t)i * 4 + j) * N_ + n) * V_ + tid;
    Q[base] = sq[tid] * (1.f / (8.f * T_));
    Kv[base] = sk[tid] * (1.f / (8.f * T_));
  }
}

// ---------------- K5: attention maps (relu-lin + outer product + softmax) -----------------
__global__ void k5_maps(const float* __restrict__ Q, const float* __restrict__ Kv,
                        const float* __restrict__ lin_w, const float* __restrict__ lin_b,
                        const float* __restrict__ att_w, const float* __restrict__ att_b,
                        float* __restrict__ maps) {
  __shared__ float qv[V_], kv[V_], ql[V_], kl[V_];
  int tid = threadIdx.x;
  int i = blockIdx.x >> 8;
  int rem = blockIdx.x & 255;
  int jm = rem >> 6, n = rem & 63;
  const float *qsrc, *ksrc, *w, *b;
  if (jm == 0) {
    qsrc = Q + (((size_t)i * 4 + 0) * N_ + n) * V_;
    ksrc = Kv + (((size_t)i * 4 + 0) * N_ + n) * V_;
    w = att_w; b = att_b;
  } else {
    qsrc = Kv + (((size_t)i * 4 + (jm - 1)) * N_ + n) * V_;
    ksrc = Q + (((size_t)i * 4 + jm) * N_ + n) * V_;
    w = lin_w; b = lin_b;
  }
  if (tid < V_) { qv[tid] = qsrc[tid]; kv[tid] = ksrc[tid]; }
  __syncthreads();
  if (tid < V_) {
    float sq = b[tid], sk = b[tid];
    for (int u = 0; u < V_; ++u) {
      sq += w[tid * V_ + u] * qv[u];
      sk += w[tid * V_ + u] * kv[u];
    }
    ql[tid] = fmaxf(sq, 0.f);
    kl[tid] = fmaxf(sk, 0.f);
  }
  __syncthreads();
  if (tid < V_) {
    float q = ql[tid];
    float mx = -1e30f;
    for (int u = 0; u < V_; ++u) mx = fmaxf(mx, q * kl[u]);
    float ex[V_], sum = 0.f;
    for (int u = 0; u < V_; ++u) { ex[u] = __expf(q * kl[u] - mx); sum += ex[u]; }
    float inv = 1.f / sum;
    size_t base = (((size_t)i * 4 + jm) * N_ + n) * (V_ * V_) + (size_t)tid * V_;
    for (int u = 0; u < V_; ++u) maps[base + u] = ex[u] * inv;
  }
}

// ---------------- K7: KNN edge conv -> edge3[n,o,v] ----------------
__global__ void k7_edge(const float* __restrict__ xm,
                        const float* __restrict__ ecw, const float* __restrict__ ecg,
                        const float* __restrict__ ecb, const float* __restrict__ edw,
                        const float* __restrict__ edb, float* __restrict__ edge3) {
  __shared__ float xe[CIN][V_];
  __shared__ float xx[V_];
  __shared__ float pd[V_][V_];
  __shared__ int knn_i[V_][KNN];
  __shared__ float emax[COUT][V_];
  int tid = threadIdx.x, n = blockIdx.x;
  for (int flat = tid; flat < CIN * V_; flat += 256)
    xe[flat / V_][flat % V_] = xm[(size_t)n * CIN * V_ + flat];
  __syncthreads();
  if (tid < V_) {
    float s = 0.f;
    for (int c = 0; c < CIN; ++c) s += xe[c][tid] * xe[c][tid];
    xx[tid] = s;
  }
  __syncthreads();
  for (int flat = tid; flat < V_ * V_; flat += 256) {
    int v = flat / V_, w = flat % V_;
    float d = 0.f;
    for (int c = 0; c < CIN; ++c) d += xe[c][v] * xe[c][w];
    pd[v][w] = 2.f * d - xx[v] - xx[w];
  }
  __syncthreads();
  if (tid < V_) {
    unsigned used = 0u;
    for (int k = 0; k < KNN; ++k) {
      float best = -3.4e38f; int bi = 0;
      for (int w = 0; w < V_; ++w)
        if (!((used >> w) & 1u) && pd[tid][w] > best) { best = pd[tid][w]; bi = w; }
      used |= (1u << bi);
      knn_i[tid][k] = bi;
    }
  }
  __syncthreads();
  const float invs = rsqrtf(1.f + EPS_);
  for (int flat = tid; flat < COUT * V_; flat += 256) {
    int o = flat / V_, v = flat % V_;
    float best = -3.4e38f;
    for (int k = 0; k < KNN; ++k) {
      int u = knn_i[v][k];
      float s = 0.f;
      for (int c = 0; c < CIN; ++c) s += ecw[o * 2 * CIN + c] * (xe[c][u] - xe[c][v]);
      for (int c = 0; c < CIN; ++c) s += ecw[o * 2 * CIN + CIN + c] * xe[c][v];
      s = s * invs * ecg[o] + ecb[o];
      s = s > 0.f ? s : 0.2f * s;
      best = fmaxf(best, s);
    }
    emax[o][v] = best;
  }
  __syncthreads();
  for (int flat = tid; flat < COUT * V_; flat += 256) {
    int o = flat / V_, v = flat % V_;
    float s = edb[o];
    for (int c = 0; c < COUT; ++c) s += edw[o * COUT + c] * emax[c][v];
    edge3[((size_t)n * COUT + o) * V_ + v] = s;
  }
}

// ---------------- K6: y = sum_i xi @ A_fin (WMMA), fused BN+edge+residual+relu ------------
__global__ void k6_y_final(const _Float16* __restrict__ xws,
                           const float* __restrict__ x0,
                           const float* __restrict__ A_base, const float* __restrict__ pre_def,
                           const float* __restrict__ alpha,
                           const float* __restrict__ maps, const float* __restrict__ acorr,
                           const float* __restrict__ edge3,
                           const float* __restrict__ bn_g, const float* __restrict__ bn_b,
                           float* __restrict__ out) {
  __shared__ _Float16 Xf[T_][VP];   // [t][v] contiguous 16KB tile (pads may hold garbage)
  __shared__ _Float16 Bt[VP][VP];   // Bt[w][v] = A_fin[v][w]; rows/cols >= 25 are zero
  int tid = threadIdx.x;
  int n = blockIdx.x >> 6, c = blockIdx.x & 63;
  const int wave = tid >> 5, lane = tid & 31, half = lane >> 4, l16 = lane & 15;
  float a0 = alpha[0];
  v8f acc[4];
#pragma unroll
  for (int q = 0; q < 4; ++q)
#pragma unroll
    for (int r = 0; r < 8; ++r) acc[q][r] = 0.f;

  _Float16* XfFlat = &Xf[0][0];
  int g = c & 7;      // tile -> group = c % GROUPS
  int jm = c >> 4;    // repeat(maps,16) -> map = c / 16

  for (int i = 0; i < NSUB; ++i) {
    __syncthreads();   // previous iteration's fragment reads complete
    const _Float16* xp = xws + ((size_t)n * MTOT + i * COUT + c) * COLSP;

#if ASYNC_LDS
    // Async DMA: 16KB contiguous global -> LDS, 16B per lane (512B per wave) per instruction, no VGPR staging.
    {
      auto gp = (__attribute__((address_space(1))) char*)(const void*)xp;
      auto lp = (__attribute__((address_space(3))) char*)XfFlat;
      int base = (wave * 16) * 128 + lane * 16;
#pragma unroll
      for (int itx = 0; itx < 16; ++itx) {
        int off = base + itx * 128;
        __builtin_amdgcn_global_load_async_to_lds_b128(
            (__attribute__((address_space(1))) i32x4*)(gp + off),
            (__attribute__((address_space(3))) i32x4*)(lp + off), 0, 0);
      }
    }
#else
    for (int flat = tid; flat < T_ * VP; flat += 256) XfFlat[flat] = xp[flat];
#endif

    // Adjacency tile (single pass: value or zero-pad) overlaps the async copy.
    for (int flat = tid; flat < VP * VP; flat += 256) {
      int w = flat >> 5, v = flat & 31;
      float val = 0.f;
      if (v < V_ && w < V_) {
        size_t gi = (((size_t)i * 8 + g) * V_ + v) * V_ + w;
        int vw = v * V_ + w;
        val = A_base[gi] + pre_def[gi]
            + a0 * maps[(((size_t)i * 4 + jm) * N_ + n) * (V_ * V_) + vw]
            + acorr[vw];
      }
      Bt[w][v] = (_Float16)val;
    }

#if ASYNC_LDS
    __builtin_amdgcn_s_wait_asynccnt(0);   // this wave's DMA complete
#endif
    __syncthreads();                       // all waves' fills visible

#pragma unroll
    for (int q = 0; q < 4; ++q) {
      int tile = wave * 4 + q;             // 16 t-tiles x 2 w-tiles
      int t0 = (tile >> 1) * 16;
      int w0 = (tile & 1) * 16;
      v16h a = wmma_pack_a(&Xf[t0 + l16][0], 0, half);
      v16h b = wmma_pack_a(&Bt[w0 + l16][0], 0, half);
      acc[q] = __builtin_amdgcn_wmma_f32_16x16x32_f16(false, a, false, b, (short)0, acc[q],
                                                      false, false);
    }
  }

  float s = bn_g[c] * rsqrtf(1.f + EPS_);
  float bb = bn_b[c];
#pragma unroll
  for (int q = 0; q < 4; ++q) {
    int tile = wave * 4 + q;
    int t0 = (tile >> 1) * 16;
    int w0 = (tile & 1) * 16;
    int vcol = w0 + l16;
    if (vcol < V_) {
      float ev = edge3[((size_t)n * COUT + c) * V_ + vcol];
#pragma unroll
      for (int r = 0; r < 8; ++r) {
        int t = t0 + r + 8 * half;
        size_t oi = (((size_t)n * COUT + c) * T_ + t) * V_ + vcol;
        float z = (acc[q][r] + ev) * s + bb + x0[oi];
        out[oi] = fmaxf(z, 0.f);
      }
    }
  }
}

extern "C" void kernel_launch(void* const* d_in, const int* in_sizes, int n_in,
                              void* d_out, int out_size, void* d_ws, size_t ws_size,
                              hipStream_t stream) {
  (void)in_sizes; (void)n_in; (void)out_size; (void)ws_size;
  const float* x0      = (const float*)d_in[0];
  const float* conv_w  = (const float*)d_in[1];
  const float* conv_b  = (const float*)d_in[2];
  const float* pre_def = (const float*)d_in[3];
  const float* A_base  = (const float*)d_in[4];
  const float* alpha   = (const float*)d_in[5];
  const float* lin_w   = (const float*)d_in[6];
  const float* lin_b   = (const float*)d_in[7];
  const float* att_w   = (const float*)d_in[8];
  const float* att_b   = (const float*)d_in[9];
  const float* cs_w    = (const float*)d_in[10];
  const float* cs_b    = (const float*)d_in[11];
  const float* cs_a    = (const float*)d_in[12];
  const float* ec_w    = (const float*)d_in[13];
  const float* ec_g    = (const float*)d_in[14];
  const float* ec_b    = (const float*)d_in[15];
  const float* ed_w    = (const float*)d_in[16];
  const float* ed_b    = (const float*)d_in[17];
  const float* bn_g    = (const float*)d_in[18];
  const float* bn_b    = (const float*)d_in[19];
  float* out = (float*)d_out;

  _Float16* xws = (_Float16*)d_ws;                  // padded f16 conv output (N x 192 x 8192)
  float* fp = (float*)(xws + (size_t)N_ * MTOT * COLSP);
  float* xm = fp;  fp += (size_t)N_ * CIN * V_;
  float* h  = fp;  fp += (size_t)N_ * COUT * V_;
  float* ac = fp;  fp += 640;
  float* Q  = fp;  fp += (size_t)NSUB * 4 * N_ * V_;
  float* Kv = fp;  fp += (size_t)NSUB * 4 * N_ * V_;
  float* mp = fp;  fp += (size_t)NSUB * 4 * N_ * V_ * V_;
  float* e3 = fp;  fp += (size_t)N_ * COUT * V_;

  k1_mean_t<<<N_ * CIN, 256, 0, stream>>>(x0, xm);
  k2a_h<<<N_, 256, 0, stream>>>(xm, cs_w, cs_b, h);
  k2b_acorr<<<V_ * V_, 256, 0, stream>>>(h, cs_a, ac);
  k3_conv_wmma<<<N_ * 128, 256, 0, stream>>>(x0, conv_w, conv_b, xws);
  k4_qk<<<N_ * 12, 256, 0, stream>>>(xws, Q, Kv);
  k5_maps<<<NSUB * 256, 32, 0, stream>>>(Q, Kv, lin_w, lin_b, att_w, att_b, mp);
  k7_edge<<<N_, 256, 0, stream>>>(xm, ec_w, ec_g, ec_b, ed_w, ed_b, e3);
  k6_y_final<<<N_ * COUT, 256, 0, stream>>>(xws, x0, A_base, pre_def, alpha, mp, ac, e3,
                                            bn_g, bn_b, out);
}
